// CausalSelfAttention_36223754175049
// MI455X (gfx1250) — compile-verified, matrix operand order checked by reading, not run on hardware
//
#include <hip/hip_runtime.h>
#include <hip/hip_bf16.h>

// ---------------------------------------------------------------------------
// CausalSelfAttention forward for MI455X (gfx1250), wave32 + WMMA f16.
// B=8, T=1024, C=1024, H=16, hd=64.  out = (y[B,T,C], k[B,H,T,hd], v[B,H,T,hd])
// ---------------------------------------------------------------------------

typedef __attribute__((ext_vector_type(16))) _Float16 v16h;
typedef __attribute__((ext_vector_type(8)))  _Float16 v8h;
typedef __attribute__((ext_vector_type(8)))  float    v8f;

#define CB   8
#define CT   1024
#define CC   1024
#define CH   16
#define CHD  64
#define CM   (CB * CT)          // 8192 rows

#define LDS_K      32
#define LDS_PAD    8
#define LDS_STRIDE (LDS_K + LDS_PAD)   // 40 halves = 80 bytes (16B aligned rows)

// ---------------------------------------------------------------------------
// fp32 -> fp16 convert
// ---------------------------------------------------------------------------
__global__ __launch_bounds__(256)
void f32_to_f16_kernel(const float* __restrict__ in, _Float16* __restrict__ out,
                       size_t n) {
    size_t i = (size_t)blockIdx.x * blockDim.x + threadIdx.x;
    if (i < n) out[i] = (_Float16)in[i];
}

// ---------------------------------------------------------------------------
// weight_norm (dim=0): Wh[o,k] = f16( g[o] * V[o,k] / ||V[o,:]|| )
// one block per output row o
// ---------------------------------------------------------------------------
__global__ __launch_bounds__(256)
void weight_norm_kernel(const float* __restrict__ V, const float* __restrict__ g,
                        _Float16* __restrict__ Wh, int K) {
    const int o = blockIdx.x;
    __shared__ float red[8];
    float ss = 0.0f;
    for (int k = threadIdx.x; k < K; k += 256) {
        float w = V[(size_t)o * K + k];
        ss += w * w;
    }
    #pragma unroll
    for (int m = 1; m < 32; m <<= 1) ss += __shfl_xor(ss, m, 32);
    if ((threadIdx.x & 31) == 0) red[threadIdx.x >> 5] = ss;
    __syncthreads();
    float tot = 0.0f;
    #pragma unroll
    for (int i = 0; i < 8; ++i) tot += red[i];
    const float scale = g[o] * rsqrtf(tot + 1e-30f);
    for (int k = threadIdx.x; k < K; k += 256)
        Wh[(size_t)o * K + k] = (_Float16)(V[(size_t)o * K + k] * scale);
}

// ---------------------------------------------------------------------------
// WMMA GEMM:  C[M,N] = A[M,K] * Bw[N,K]^T   (both f16 row-major, f32 accum)
// Block tile 128x128, K-step 32. 8 waves arranged 4(M) x 2(N), wave tile 32x64.
// WRITE_F32 = 0: write f16 to Ch;  1: write f32 to Cf.
// ---------------------------------------------------------------------------
template <int WRITE_F32>
__global__ __launch_bounds__(256)
void wmma_gemm_nt(const _Float16* __restrict__ A,
                  const _Float16* __restrict__ Bw,
                  _Float16* __restrict__ Ch,
                  float* __restrict__ Cf,
                  int M, int N, int K) {
    __shared__ _Float16 As[128 * LDS_STRIDE];
    __shared__ _Float16 Bs[128 * LDS_STRIDE];

    const int tid  = threadIdx.x;
    const int wave = tid >> 5;
    const int lane = tid & 31;
    const int l16  = lane & 15;
    const int half = lane >> 4;
    const int koff = half * 8;          // A/B fragment K-chunk base per half-wave

    const int bm = blockIdx.x * 128;
    const int bn = blockIdx.y * 128;
    const int wm = (wave & 3) * 32;     // wave M offset in block
    const int wn = (wave >> 2) * 64;    // wave N offset in block

    v8f acc[2][4] = {};

    for (int k0 = 0; k0 < K; k0 += LDS_K) {
        // stage A/B tiles (128 rows x 32 halves) via 16B chunks
        #pragma unroll
        for (int it = 0; it < 2; ++it) {
            int i = tid + it * 256;                 // 0..511
            int r = i >> 2;
            int c = (i & 3) * 8;
            *(v8h*)&As[r * LDS_STRIDE + c] =
                *(const v8h*)&A[(size_t)(bm + r) * K + k0 + c];
            *(v8h*)&Bs[r * LDS_STRIDE + c] =
                *(const v8h*)&Bw[(size_t)(bn + r) * K + k0 + c];
        }
        __syncthreads();

        if (k0 + LDS_K < K) {   // gfx1250 global_prefetch_b8 on next K tile
            __builtin_prefetch(&A[(size_t)(bm + (tid >> 1)) * K + k0 + LDS_K], 0, 1);
            __builtin_prefetch(&Bw[(size_t)(bn + (tid >> 1)) * K + k0 + LDS_K], 0, 1);
        }

        // A fragments (2 M-tiles): lanes 0-15 row=l16 K{0-7,16-23}, lanes 16-31 K{8-15,24-31}
        v16h af[2];
        #pragma unroll
        for (int mt = 0; mt < 2; ++mt) {
            const _Float16* p = &As[(wm + mt * 16 + l16) * LDS_STRIDE + koff];
            v8h lo = *(const v8h*)p;
            v8h hi = *(const v8h*)(p + 16);
            #pragma unroll
            for (int i = 0; i < 8; ++i) { af[mt][i] = lo[i]; af[mt][i + 8] = hi[i]; }
        }
        // B fragments (4 N-tiles), mirrored layout (col = l16, same K chunks)
        v16h bf[4];
        #pragma unroll
        for (int nt = 0; nt < 4; ++nt) {
            const _Float16* p = &Bs[(wn + nt * 16 + l16) * LDS_STRIDE + koff];
            v8h lo = *(const v8h*)p;
            v8h hi = *(const v8h*)(p + 16);
            #pragma unroll
            for (int i = 0; i < 8; ++i) { bf[nt][i] = lo[i]; bf[nt][i + 8] = hi[i]; }
        }
        #pragma unroll
        for (int mt = 0; mt < 2; ++mt)
            #pragma unroll
            for (int nt = 0; nt < 4; ++nt)
                acc[mt][nt] = __builtin_amdgcn_wmma_f32_16x16x32_f16(
                    false, af[mt], false, bf[nt], (short)0, acc[mt][nt],
                    false, false);
        __syncthreads();
    }

    // epilogue: C layout -> row = r + 8*half, col = l16
    #pragma unroll
    for (int mt = 0; mt < 2; ++mt)
        #pragma unroll
        for (int nt = 0; nt < 4; ++nt)
            #pragma unroll
            for (int r = 0; r < 8; ++r) {
                int row = bm + wm + mt * 16 + r + half * 8;
                int col = bn + wn + nt * 16 + l16;
                if (WRITE_F32)
                    Cf[(size_t)row * N + col] = acc[mt][nt][r];
                else
                    Ch[(size_t)row * N + col] = (_Float16)acc[mt][nt][r];
            }
}

// ---------------------------------------------------------------------------
// k/v finalize: f32 outputs in [B,H,T,hd] + f16 V^T [B,H,hd,T] for attention
// ---------------------------------------------------------------------------
__global__ __launch_bounds__(256)
void kv_finalize(const _Float16* __restrict__ kh, const _Float16* __restrict__ vh,
                 float* __restrict__ kout, float* __restrict__ vout,
                 _Float16* __restrict__ vT) {
    size_t i = (size_t)blockIdx.x * 256 + threadIdx.x;   // over B*T*C
    if (i >= (size_t)CM * CC) return;
    int c = (int)(i % CC);
    size_t bt = i / CC;
    int t = (int)(bt % CT);
    int b = (int)(bt / CT);
    int h = c >> 6, d = c & 63;
    size_t o = (((size_t)b * CH + h) * CT + t) * CHD + d;
    kout[o] = (float)kh[i];
    _Float16 vv = vh[i];
    vout[o] = (float)vv;
    vT[(((size_t)b * CH + h) * CHD + d) * CT + t] = vv;
}

// ---------------------------------------------------------------------------
// Flash attention: grid (B*H, T/128), 8 waves x 16 query rows each.
// 32 keys per iteration; online softmax; all matmuls via WMMA f16.
// ---------------------------------------------------------------------------
__global__ __launch_bounds__(256)
void attn_kernel(const _Float16* __restrict__ qh, const _Float16* __restrict__ kh,
                 const _Float16* __restrict__ vT, _Float16* __restrict__ yh) {
    const int bh   = blockIdx.x;           // b*H + h
    const int b    = bh >> 4;
    const int h    = bh & 15;
    const int wave = threadIdx.x >> 5;
    const int lane = threadIdx.x & 31;
    const int l16  = lane & 15;
    const int half = lane >> 4;
    const int koff = half * 8;
    const int qrow0 = blockIdx.y * 128 + wave * 16;   // 16 query rows per wave

    __shared__ _Float16 Pst[8][16 * 40];  // per-wave P staging, 80B row stride

    // preload Q A-fragments for K-dim 0..63 (2 frags of 16x32)
    v16h qf[2];
    {
        const size_t qbase = ((size_t)(b * CT + qrow0 + l16)) * CC + h * CHD;
        #pragma unroll
        for (int kk = 0; kk < 2; ++kk) {
            const _Float16* p = &qh[qbase + kk * 32 + koff];
            v8h lo = *(const v8h*)p;
            v8h hi = *(const v8h*)(p + 16);
            #pragma unroll
            for (int i = 0; i < 8; ++i) { qf[kk][i] = lo[i]; qf[kk][i + 8] = hi[i]; }
        }
    }

    v8f   o_acc[4] = {};         // output tiles d0 = 0,16,32,48
    float mi[8], li[8];
    #pragma unroll
    for (int r = 0; r < 8; ++r) { mi[r] = -1e30f; li[r] = 0.0f; }

    const int   qmax   = qrow0 + 15;
    const float sm     = 0.125f;                       // 1/sqrt(64)
    const size_t krow0 = (size_t)b * CT;
    const size_t vtb   = (size_t)bh * CHD * CT;

    for (int j0 = 0; j0 <= qmax; j0 += 32) {
        // ---- S = Q * K^T for keys [j0, j0+32) : two 16x16 tiles -----------
        v8f s[2] = {};
        #pragma unroll
        for (int jt = 0; jt < 2; ++jt) {
            const _Float16* kp =
                &kh[(krow0 + j0 + jt * 16 + l16) * CC + h * CHD + koff];
            #pragma unroll
            for (int kk = 0; kk < 2; ++kk) {
                v16h bfr;
                v8h lo = *(const v8h*)(kp + kk * 32);
                v8h hi = *(const v8h*)(kp + kk * 32 + 16);
                #pragma unroll
                for (int i = 0; i < 8; ++i) { bfr[i] = lo[i]; bfr[i + 8] = hi[i]; }
                s[jt] = __builtin_amdgcn_wmma_f32_16x16x32_f16(
                    false, qf[kk], false, bfr, (short)0, s[jt], false, false);
            }
        }

        // ---- scale + causal mask + online softmax -------------------------
        #pragma unroll
        for (int r = 0; r < 8; ++r) {
            const int qi  = qrow0 + r + half * 8;
            float v0 = s[0][r] * sm;
            float v1 = s[1][r] * sm;
            if (j0 + l16 > qi)      v0 = -1e30f;
            if (j0 + 16 + l16 > qi) v1 = -1e30f;

            float mloc = fmaxf(v0, v1);
            #pragma unroll
            for (int mk = 1; mk < 16; mk <<= 1)
                mloc = fmaxf(mloc, __shfl_xor(mloc, mk, 16));
            const float mnew  = fmaxf(mi[r], mloc);
            const float scale = __expf(mi[r] - mnew);
            const float p0 = __expf(v0 - mnew);
            const float p1 = __expf(v1 - mnew);
            float rs = p0 + p1;
            #pragma unroll
            for (int mk = 1; mk < 16; mk <<= 1) rs += __shfl_xor(rs, mk, 16);
            li[r] = li[r] * scale + rs;
            mi[r] = mnew;
            #pragma unroll
            for (int nt = 0; nt < 4; ++nt) o_acc[nt][r] *= scale;

            // stage P (row = r + 8*half, cols l16 / 16+l16) for A-frag reload
            Pst[wave][(r + half * 8) * 40 + l16]      = (_Float16)p0;
            Pst[wave][(r + half * 8) * 40 + 16 + l16] = (_Float16)p1;
        }

        // ---- reload P as 16x32 A-fragment (wave-local LDS is in-order) ----
        v16h pf;
        {
            const _Float16* pp = &Pst[wave][l16 * 40 + koff];
            v8h lo = *(const v8h*)pp;
            v8h hi = *(const v8h*)(pp + 16);
            #pragma unroll
            for (int i = 0; i < 8; ++i) { pf[i] = lo[i]; pf[i + 8] = hi[i]; }
        }

        // ---- O += P * V  (V^T layout -> contiguous along keys) ------------
        #pragma unroll
        for (int nt = 0; nt < 4; ++nt) {
            const _Float16* vp = &vT[vtb + (size_t)(nt * 16 + l16) * CT + j0 + koff];
            v16h bfr;
            v8h lo = *(const v8h*)vp;
            v8h hi = *(const v8h*)(vp + 16);
            #pragma unroll
            for (int i = 0; i < 8; ++i) { bfr[i] = lo[i]; bfr[i + 8] = hi[i]; }
            o_acc[nt] = __builtin_amdgcn_wmma_f32_16x16x32_f16(
                false, pf, false, bfr, (short)0, o_acc[nt], false, false);
        }
    }

    // ---- normalize + store y (f16 workspace, [B*T, C]) --------------------
    #pragma unroll
    for (int nt = 0; nt < 4; ++nt)
        #pragma unroll
        for (int r = 0; r < 8; ++r) {
            int row = qrow0 + r + half * 8;
            int d   = nt * 16 + l16;
            yh[((size_t)(b * CT + row)) * CC + h * CHD + d] =
                (_Float16)(o_acc[nt][r] / li[r]);
        }
}

// ---------------------------------------------------------------------------
// host launch
// ---------------------------------------------------------------------------
extern "C" void kernel_launch(void* const* d_in, const int* in_sizes, int n_in,
                              void* d_out, int out_size, void* d_ws, size_t ws_size,
                              hipStream_t stream) {
    const float* x   = (const float*)d_in[0];
    const float* Wqv = (const float*)d_in[1];
    const float* gq  = (const float*)d_in[2];
    const float* Wkv = (const float*)d_in[3];
    const float* gk  = (const float*)d_in[4];
    const float* Wv  = (const float*)d_in[5];
    const float* Wc  = (const float*)d_in[6];
    (void)in_sizes; (void)n_in; (void)out_size; (void)ws_size;

    const size_t NXC = (size_t)CM * CC;     // 8388608
    const size_t NW  = (size_t)CC * CC;     // 1048576

    float* y_out = (float*)d_out;
    float* k_out = y_out + NXC;
    float* v_out = k_out + NXC;

    _Float16* xh  = (_Float16*)d_ws;
    _Float16* Wqh = xh  + NXC;
    _Float16* Wkh = Wqh + NW;
    _Float16* Wvh = Wkh + NW;
    _Float16* Wch = Wvh + NW;
    _Float16* qh  = Wch + NW;
    _Float16* kh  = qh  + NXC;
    _Float16* vh  = kh  + NXC;
    _Float16* vTh = vh  + NXC;
    _Float16* yh  = vTh + NXC;

    // 1) convert x, Wv, Wc; weight-norm Wq, Wk (fused with convert)
    f32_to_f16_kernel<<<dim3((unsigned)(NXC / 256)), 256, 0, stream>>>(x, xh, NXC);
    f32_to_f16_kernel<<<dim3((unsigned)(NW / 256)), 256, 0, stream>>>(Wv, Wvh, NW);
    f32_to_f16_kernel<<<dim3((unsigned)(NW / 256)), 256, 0, stream>>>(Wc, Wch, NW);
    weight_norm_kernel<<<CC, 256, 0, stream>>>(Wqv, gq, Wqh, CC);
    weight_norm_kernel<<<CC, 256, 0, stream>>>(Wkv, gk, Wkh, CC);

    // 2) QKV projections: [8192,1024] x [1024,1024]^T
    dim3 gg(CM / 128, CC / 128);
    wmma_gemm_nt<0><<<gg, 256, 0, stream>>>(xh, Wqh, qh, nullptr, CM, CC, CC);
    wmma_gemm_nt<0><<<gg, 256, 0, stream>>>(xh, Wkh, kh, nullptr, CM, CC, CC);
    wmma_gemm_nt<0><<<gg, 256, 0, stream>>>(xh, Wvh, vh, nullptr, CM, CC, CC);

    // 3) k/v fp32 outputs ([B,H,T,hd]) + V^T for attention
    kv_finalize<<<dim3((unsigned)(NXC / 256)), 256, 0, stream>>>(kh, vh, k_out,
                                                                 v_out, vTh);

    // 4) flash attention
    attn_kernel<<<dim3(CB * CH, CT / 128), 256, 0, stream>>>(qh, kh, vTh, yh);

    // 5) output projection -> fp32 y
    wmma_gemm_nt<1><<<gg, 256, 0, stream>>>(yh, Wch, nullptr, y_out, CM, CC, CC);
}
